// NormalSeq2SeqModel_4939212390896
// MI455X (gfx1250) — matrix operand order repaired and checked by reading, not compiled, not run on hardware
//
#include <hip/hip_runtime.h>
#include <hip/hip_bf16.h>
#include <stdint.h>

typedef __bf16 bf16_t;
typedef __attribute__((ext_vector_type(16))) __bf16 v16bf;
typedef __attribute__((ext_vector_type(8)))  __bf16 v8bf;
typedef __attribute__((ext_vector_type(8)))  float  v8f;

namespace {
constexpr int B_     = 256;
constexpr int S_     = 1024;
constexpr int H_     = 512;
constexpr int G4_    = 4 * H_;      // 2048 gate rows
constexpr int TLEN   = 256;
constexpr int TSTEPS = S_ + TLEN;   // 1280 sequential steps
constexpr int JT     = 32;          // h-columns (per gate) owned by a block
constexpr int NBT    = B_ / 16;     // 16 batch tiles of 16 rows
constexpr int NJT    = H_ / JT;     // 16 j tiles
constexpr int WROWS  = 4 * JT;      // 128 gate rows resident in LDS
constexpr int WPITCH = H_ + 8;      // halves per LDS weight row (pad 16B: bank spread)
constexpr int HPITCH = H_ + 8;      // halves per LDS h row
constexpr size_t LDS_BYTES =
    (size_t)WROWS * WPITCH * 2 + (size_t)16 * HPITCH * 2 + 64;  // ~150 KB
}

__device__ __forceinline__ float sigm(float x) { return 1.0f / (1.0f + __expf(-x)); }
__device__ __forceinline__ float tanh_(float x) {
  float e = __expf(2.0f * x);
  return (e - 1.0f) / (e + 1.0f);
}
__device__ __forceinline__ v16bf cat16(v8bf lo, v8bf hi) {
  return __builtin_shufflevector(lo, hi, 0, 1, 2, 3, 4, 5, 6, 7, 8, 9, 10, 11, 12, 13, 14, 15);
}

// Copy this block's 128x512 bf16 weight slab (gate rows g*H + j0 + jj) into LDS.
__device__ __forceinline__ void load_w_slab(bf16_t* w_lds, const bf16_t* __restrict__ wsrc,
                                            int j0, int tid) {
  for (int i = tid; i < WROWS * (H_ / 8); i += 64) {
    const int row = i >> 6;   // LDS row 0..127
    const int c8  = i & 63;   // 16B chunk within row
    const int g   = row >> 5;
    const int jj  = row & 31;
    const uint4 v = *(const uint4*)(wsrc + (size_t)(g * H_ + j0 + jj) * H_ + c8 * 8);
    *(uint4*)(w_lds + row * WPITCH + c8 * 8) = v;
  }
}

__global__ void k_convert(const float* __restrict__ ew, const float* __restrict__ dw,
                          bf16_t* __restrict__ eb, bf16_t* __restrict__ db, int n) {
  const int i = blockIdx.x * blockDim.x + threadIdx.x;
  if (i < n) {
    eb[i] = (bf16_t)ew[i];
    db[i] = (bf16_t)dw[i];
  }
}

__global__ void k_zero(uint32_t* a, int na, uint32_t* b, int nb) {
  const int stride = gridDim.x * blockDim.x;
  for (int j = blockIdx.x * blockDim.x + threadIdx.x; j < na; j += stride) a[j] = 0u;
  for (int j = blockIdx.x * blockDim.x + threadIdx.x; j < nb; j += stride) b[j] = 0u;
}

// Persistent seq2seq LSTM. Grid: 256 blocks = 16 batch-tiles x 16 j-tiles, 64 threads (2 waves).
// Wave w handles N-subtile w (16 gate columns) for all 4 gates -> elementwise update is lane-local.
__global__ __launch_bounds__(64) void lstm_persistent(
    const float* __restrict__ inputs,
    const float* __restrict__ enc_wih, const float* __restrict__ enc_bih,
    const float* __restrict__ enc_bhh,
    const float* __restrict__ dec_wih, const float* __restrict__ dec_bih,
    const float* __restrict__ dec_bhh,
    const float* __restrict__ linW, const float* __restrict__ linb,
    const bf16_t* __restrict__ enc_wbf, const bf16_t* __restrict__ dec_wbf,
    bf16_t* __restrict__ hbuf0, bf16_t* __restrict__ hbuf1,
    unsigned* __restrict__ flags, float* __restrict__ out) {
  extern __shared__ char smem[];
  bf16_t* w_lds = (bf16_t*)smem;                 // 128 x WPITCH halves
  bf16_t* h_lds = w_lds + WROWS * WPITCH;        // 16 x HPITCH halves
  float*  x_lds = (float*)(h_lds + 16 * HPITCH); // 16 floats

  const int bt   = blockIdx.x & (NBT - 1);
  const int jt   = blockIdx.x / NBT;
  const int b0   = bt * 16;
  const int j0   = jt * JT;
  const int tid  = threadIdx.x;
  const int wv   = tid >> 5;   // wave index 0/1 (wave32)
  const int lane = tid & 31;
  const int lhalf = lane >> 4; // 0: lanes 0-15, 1: lanes 16-31
  const int l16  = lane & 15;
  const int jj   = wv * 16 + l16;  // column within block's 32
  const int col  = j0 + jj;        // 0..511 (per-gate h column)

  // per-lane constants: biases, rank-1 input weights, output projection
  float be[4], bd[4], we[4], wd[4];
#pragma unroll
  for (int g = 0; g < 4; ++g) {
    be[g] = enc_bih[g * H_ + col] + enc_bhh[g * H_ + col];
    bd[g] = dec_bih[g * H_ + col] + dec_bhh[g * H_ + col];
    we[g] = enc_wih[g * H_ + col];
    wd[g] = dec_wih[g * H_ + col];
  }
  const float lw = linW[col];
  const float lb = linb[0];
  unsigned* flg = flags + bt * 16;  // one cacheline per batch tile

  load_w_slab(w_lds, enc_wbf, j0, tid);  // encoder weights resident in LDS

  // Per-lane fragment base pointers (constant for the whole run; inner loop
  // addresses are base + kb*64B immediates -> ds_load_b128 with imm offsets).
  // A layout: lanes 0-15 hold K{0-7,16-23}, lanes 16-31 hold K{8-15,24-31}, row = l16.
  // B layout: lanes 0-15 hold K{0-7,16-23}, lanes 16-31 hold K{8-15,24-31}, column = l16.
  const int koffA = lhalf ? 8 : 0;
  const int koffB = lhalf ? 8 : 0;
  const bf16_t* aBase = h_lds + l16 * HPITCH + koffA;
  const bf16_t* bBase0 = w_lds + (0 * JT + jj) * WPITCH + koffB;
  const bf16_t* bBase1 = w_lds + (1 * JT + jj) * WPITCH + koffB;
  const bf16_t* bBase2 = w_lds + (2 * JT + jj) * WPITCH + koffB;
  const bf16_t* bBase3 = w_lds + (3 * JT + jj) * WPITCH + koffB;

  auto loadA = [&](int kb) -> v16bf {
    const bf16_t* p = aBase + kb * 32;
    return cat16(*(const v8bf*)p, *(const v8bf*)(p + 16));
  };
  auto loadB = [&](const bf16_t* base, int kb) -> v16bf {
    const bf16_t* p = base + kb * 32;
    return cat16(*(const v8bf*)p, *(const v8bf*)(p + 16));
  };

  float cst[8];  // cell state, fp32, persists across all 1280 steps
#pragma unroll
  for (int r = 0; r < 8; ++r) cst[r] = 0.0f;

  for (int t = 0; t < TSTEPS; ++t) {
    // ---- acquire: wait until all 16 j-blocks of this batch tile finished step t-1
    if (tid == 0) {
      const unsigned need = 16u * (unsigned)t;
      while (__hip_atomic_load(flg, __ATOMIC_ACQUIRE, __HIP_MEMORY_SCOPE_AGENT) < need)
        __builtin_amdgcn_s_sleep(1);
    }
    __syncthreads();
    __builtin_amdgcn_fence(__ATOMIC_ACQUIRE, "agent");

    if (t == S_) load_w_slab(w_lds, dec_wbf, j0, tid);  // swap to decoder weights

    const bf16_t* __restrict__ hread  = (t & 1) ? hbuf1 : hbuf0;  // h(t-1)
    bf16_t* __restrict__       hwrite = (t & 1) ? hbuf0 : hbuf1;  // h(t)

    // stage h tile (16 rows x 512 bf16) into LDS
    for (int i = tid; i < 16 * (H_ / 8); i += 64) {
      const int row = i >> 6;
      const int c8  = i & 63;
      const uint4 v = *(const uint4*)(hread + (size_t)(b0 + row) * H_ + c8 * 8);
      *(uint4*)(h_lds + row * HPITCH + c8 * 8) = v;
    }
    // stage x (scalar per batch row; decoder feeds back previous output)
    if (tid < 16) {
      float xv;
      if (t < S_)       xv = inputs[(size_t)(b0 + tid) * S_ + t];
      else if (t == S_) xv = inputs[(size_t)(b0 + tid) * S_ + (S_ - 1)];
      else              xv = out[(size_t)(b0 + tid) * TLEN + (t - S_ - 1)];
      x_lds[tid] = xv;
    }
    __syncthreads();

    const bool enc = (t < S_);
    v8f acc[4] = {};  // i,f,g,o fp32 accumulators, identical (m,n) layout

    // Software-pipelined GEMM: fragments for kb+1 are loaded before the four
    // WMMAs of kb issue, so ds latency overlaps the matrix pipe instead of
    // draining dscnt to 0 in front of every WMMA.
    v16bf aC  = loadA(0);
    v16bf b0C = loadB(bBase0, 0);
    v16bf b1C = loadB(bBase1, 0);
    v16bf b2C = loadB(bBase2, 0);
    v16bf b3C = loadB(bBase3, 0);
#pragma unroll
    for (int kb = 0; kb < 16; ++kb) {
      v16bf aN = {}, b0N = {}, b1N = {}, b2N = {}, b3N = {};
      if (kb < 15) {
        aN  = loadA(kb + 1);
        b0N = loadB(bBase0, kb + 1);
        b1N = loadB(bBase1, kb + 1);
        b2N = loadB(bBase2, kb + 1);
        b3N = loadB(bBase3, kb + 1);
      }
      acc[0] = __builtin_amdgcn_wmma_f32_16x16x32_bf16(false, aC, false, b0C, (short)0, acc[0], false, false);
      acc[1] = __builtin_amdgcn_wmma_f32_16x16x32_bf16(false, aC, false, b1C, (short)0, acc[1], false, false);
      acc[2] = __builtin_amdgcn_wmma_f32_16x16x32_bf16(false, aC, false, b2C, (short)0, acc[2], false, false);
      acc[3] = __builtin_amdgcn_wmma_f32_16x16x32_bf16(false, aC, false, b3C, (short)0, acc[3], false, false);
      aC = aN; b0C = b0N; b1C = b1N; b2C = b2N; b3C = b3N;
    }

    // x for the 8 rows this lane-half owns: two ds_load_b128
    float xr[8];
    {
      const float4 xlo = *(const float4*)(x_lds + 8 * lhalf);
      const float4 xhi = *(const float4*)(x_lds + 8 * lhalf + 4);
      xr[0] = xlo.x; xr[1] = xlo.y; xr[2] = xlo.z; xr[3] = xlo.w;
      xr[4] = xhi.x; xr[5] = xhi.y; xr[6] = xhi.z; xr[7] = xhi.w;
    }

    const int tdec = t - S_;
#pragma unroll
    for (int r = 0; r < 8; ++r) {
      // D layout: this element is (m = r + 8*lhalf, n = l16) of the 16x16 tile
      const float gi = acc[0][r] + (enc ? be[0] : bd[0]) + xr[r] * (enc ? we[0] : wd[0]);
      const float gf = acc[1][r] + (enc ? be[1] : bd[1]) + xr[r] * (enc ? we[1] : wd[1]);
      const float gg = acc[2][r] + (enc ? be[2] : bd[2]) + xr[r] * (enc ? we[2] : wd[2]);
      const float go = acc[3][r] + (enc ? be[3] : bd[3]) + xr[r] * (enc ? we[3] : wd[3]);
      const float cn = sigm(gf) * cst[r] + sigm(gi) * tanh_(gg);
      cst[r] = cn;
      const float hn = sigm(go) * tanh_(cn);
      const int m = r + 8 * lhalf;
      hwrite[(size_t)(b0 + m) * H_ + col] = (bf16_t)hn;
      if (!enc) {
        // partial output projection: reduce h*lw over the 16 columns this wave owns
        float p = hn * lw;
        p += __shfl_xor(p, 1, 32);
        p += __shfl_xor(p, 2, 32);
        p += __shfl_xor(p, 4, 32);
        p += __shfl_xor(p, 8, 32);
        if (l16 == 0) {
          const float addv = p + ((jt == 0 && wv == 0) ? lb : 0.0f);
          atomicAdd(&out[(size_t)(b0 + m) * TLEN + tdec], addv);
        }
      }
    }

    // ---- release: make h / out visible, then signal step completion
    __builtin_amdgcn_fence(__ATOMIC_RELEASE, "agent");
    __syncthreads();
    if (tid == 0)
      __hip_atomic_fetch_add(flg, 1u, __ATOMIC_RELEASE, __HIP_MEMORY_SCOPE_AGENT);
  }
}

extern "C" void kernel_launch(void* const* d_in, const int* in_sizes, int n_in,
                              void* d_out, int out_size, void* d_ws, size_t ws_size,
                              hipStream_t stream) {
  const float* inputs  = (const float*)d_in[0];
  const float* enc_Wih = (const float*)d_in[1];
  const float* enc_Whh = (const float*)d_in[2];
  const float* enc_bih = (const float*)d_in[3];
  const float* enc_bhh = (const float*)d_in[4];
  const float* dec_Wih = (const float*)d_in[5];
  const float* dec_Whh = (const float*)d_in[6];
  const float* dec_bih = (const float*)d_in[7];
  const float* dec_bhh = (const float*)d_in[8];
  const float* lin_W   = (const float*)d_in[9];
  const float* lin_b   = (const float*)d_in[10];
  float* out = (float*)d_out;

  // workspace carve-up (~4.7 MB)
  char* ws = (char*)d_ws;
  bf16_t* enc_wbf = (bf16_t*)ws;                         // 2048*512 bf16
  bf16_t* dec_wbf = enc_wbf + (size_t)G4_ * H_;          // 2048*512 bf16
  bf16_t* hbuf0   = dec_wbf + (size_t)G4_ * H_;          // 256*512 bf16 (ping)
  bf16_t* hbuf1   = hbuf0 + (size_t)B_ * H_;             // 256*512 bf16 (pong)
  unsigned* flags = (unsigned*)(hbuf1 + (size_t)B_ * H_);// 16 tiles * 16 uints

  // zero h exchange buffers + flags + d_out (accumulated into by decoder)
  const int nz_h = (int)(((size_t)2 * B_ * H_ * 2 + 1024) / 4);
  k_zero<<<256, 256, 0, stream>>>((uint32_t*)hbuf0, nz_h, (uint32_t*)out, B_ * TLEN);

  // convert recurrent weights to bf16 (row-major = WMMA-B friendly)
  const int nw = G4_ * H_;
  k_convert<<<(nw + 255) / 256, 256, 0, stream>>>(enc_Whh, dec_Whh, enc_wbf, dec_wbf, nw);

  lstm_persistent<<<NBT * NJT, 64, LDS_BYTES, stream>>>(
      inputs, enc_Wih, enc_bih, enc_bhh, dec_Wih, dec_bih, dec_bhh, lin_W, lin_b,
      enc_wbf, dec_wbf, hbuf0, hbuf1, flags, out);
}
